// MultiHeadAttention_17025250362019
// MI455X (gfx1250) — matrix operand order repaired and checked by reading, not compiled, not run on hardware
//
#include <hip/hip_runtime.h>
#include <hip/hip_bf16.h>
#include <math.h>

// ---------------------------------------------------------------------------
// MHA forward for gfx1250 (CDNA5): f16 WMMA GEMMs + flash attention.
// x:[4,2048,768]f32, W_qkv:[768,2304]f32, W_proj:[768,768]f32, b_proj:[768]f32
// out:[4,2048,768]f32
// ---------------------------------------------------------------------------

typedef __attribute__((ext_vector_type(16))) _Float16 v16h;
typedef __attribute__((ext_vector_type(8)))  _Float16 v8h;
typedef __attribute__((ext_vector_type(8)))  float    v8f;

#define DIM   768
#define HEADS 12
#define HD    64
#define NB    4
#define NSEQ  2048
#define BN    (NB * NSEQ)          // 8192 rows
#define KSTEPS (DIM / 32)          // 24
#define SCALE 0.036084391824351614f // 768^-0.5 (module scales by full dim)

__device__ __forceinline__ v16h cat8(v8h a, v8h b) {
  return __builtin_shufflevector(a, b, 0,1,2,3,4,5,6,7,8,9,10,11,12,13,14,15);
}

// A fragment (16x32 f16, row-major source). ISA layout: lane L (m = L&15,
// hi = L>>4) holds K = k0 + hi*8 + [0..7] in halfs 0..7 and
// K = k0 + 16 + hi*8 + [0..7] in halfs 8..15.  Two 16-byte loads.
__device__ __forceinline__ v16h load_a_frag(const _Float16* base, int ld,
                                            int k0, int lane) {
  const int m = lane & 15, hi = lane >> 4;
  const _Float16* p = base + (size_t)m * ld + k0 + hi * 8;
  v8h lo = *(const v8h*)(p);
  v8h hv = *(const v8h*)(p + 16);
  return cat8(lo, hv);
}

// B fragment (32x16 f16) from column-contiguous storage. Lane L holds column
// n = L&15; K = k0 + (L>>4)*8 + [0..7] in halfs 0..7 and
// K = k0 + 16 + (L>>4)*8 + [0..7] in halfs 8..15. Two 16-byte loads.
__device__ __forceinline__ v16h load_b_frag(const _Float16* colbase, int k0,
                                            int hi) {
  const _Float16* p = colbase + k0 + hi * 8;
  v8h lo = *(const v8h*)(p);
  v8h hv = *(const v8h*)(p + 16);
  return cat8(lo, hv);
}

__device__ __forceinline__ v8f wmma16(v16h a, v16h b, v8f c) {
  return __builtin_amdgcn_wmma_f32_16x16x32_f16(false, a, false, b,
                                                (short)0, c, false, false);
}

// ---------------------------------------------------------------------------
// Stage 0: conversions
// ---------------------------------------------------------------------------
__global__ void cvt_f32_to_f16(const float* __restrict__ src,
                               _Float16* __restrict__ dst, int n) {
  int i = blockIdx.x * blockDim.x + threadIdx.x;
  if (i < n) dst[i] = (_Float16)src[i];
}

// src [rows, cols] row-major -> dst [cols, rows] (column-contiguous weights)
__global__ void transpose_to_f16(const float* __restrict__ src,
                                 _Float16* __restrict__ dst, int rows, int cols) {
  int i = blockIdx.x * blockDim.x + threadIdx.x;
  if (i < rows * cols) {
    int r = i / cols, c = i - r * cols;
    dst[(size_t)c * rows + r] = (_Float16)src[i];
  }
}

// ---------------------------------------------------------------------------
// Stage 1: qkv = x @ W_qkv.  One wave -> 32x64 tile.  Fully-unrolled K loop
// with 2-deep compile-time ping-pong buffers (prefetch one K-step ahead, no
// register copies).  256 row tiles x 36 col tiles = 9216 waves = 1152 blocks.
// Q scaled by SCALE -> Q[B,H,N,hd]; K[B,H,N,hd]; V stored transposed
// Vt[B,H,hd,N] so stage-2 PV B-fragments are contiguous.
// ---------------------------------------------------------------------------
__global__ __launch_bounds__(256) void qkv_gemm_kernel(
    const _Float16* __restrict__ xh, const _Float16* __restrict__ wt,
    _Float16* __restrict__ Qd, _Float16* __restrict__ Kd,
    _Float16* __restrict__ Vt) {
  const int lane = threadIdx.x & 31;
  const int wid  = blockIdx.x * 8 + (threadIdx.x >> 5);
  const int rowTile = wid / 36;
  const int colT    = wid - rowTile * 36;
  const int n = lane & 15, hi = lane >> 4, m0 = hi * 8;

  const _Float16* arow0 = xh + (size_t)rowTile * 32 * DIM;
  const _Float16* arow1 = arow0 + (size_t)16 * DIM;
  const _Float16* colbase[4];
#pragma unroll
  for (int t = 0; t < 4; ++t)
    colbase[t] = wt + (size_t)(colT * 64 + t * 16 + n) * DIM;

  v8f acc[2][4] = {};
  v16h a0[2], a1[2], b[2][4];
  a0[0] = load_a_frag(arow0, DIM, 0, lane);
  a1[0] = load_a_frag(arow1, DIM, 0, lane);
#pragma unroll
  for (int t = 0; t < 4; ++t) b[0][t] = load_b_frag(colbase[t], 0, hi);

#pragma unroll
  for (int kk = 0; kk < KSTEPS; ++kk) {
    const int cur = kk & 1, nx = cur ^ 1;
    if (kk + 1 < KSTEPS) {
      const int k0 = (kk + 1) * 32;
      a0[nx] = load_a_frag(arow0, DIM, k0, lane);
      a1[nx] = load_a_frag(arow1, DIM, k0, lane);
#pragma unroll
      for (int t = 0; t < 4; ++t) b[nx][t] = load_b_frag(colbase[t], k0, hi);
    }
#pragma unroll
    for (int t = 0; t < 4; ++t) {
      acc[0][t] = wmma16(a0[cur], b[cur][t], acc[0][t]);
      acc[1][t] = wmma16(a1[cur], b[cur][t], acc[1][t]);
    }
  }

  const int which = colT / 12;   // 0=Q 1=K 2=V (64-wide tiles align to heads)
  const int h     = colT % 12;
#pragma unroll
  for (int g = 0; g < 2; ++g) {
#pragma unroll
    for (int t = 0; t < 4; ++t) {
#pragma unroll
      for (int r = 0; r < 8; ++r) {
        float v = acc[g][t][r];
        int R  = rowTile * 32 + g * 16 + m0 + r;   // global row = b*2048 + nq
        int bb = R >> 11, nq = R & 2047;
        int d  = t * 16 + n;
        if (which == 0) {
          Qd[((size_t)(bb * HEADS + h) * NSEQ + nq) * HD + d] =
              (_Float16)(v * SCALE);
        } else if (which == 1) {
          Kd[((size_t)(bb * HEADS + h) * NSEQ + nq) * HD + d] = (_Float16)v;
        } else {
          Vt[((size_t)(bb * HEADS + h) * HD + d) * NSEQ + nq] = (_Float16)v;
        }
      }
    }
  }
}

// ---------------------------------------------------------------------------
// Stage 2: flash attention.  One wave = 16 queries x full head (hd=64).
// 48 (b,h) x 128 q-tiles = 6144 waves = 768 blocks.
// Per 32-key block: 4 WMMAs (S) + online softmax + LDS bounce + 4 WMMAs (PV).
// ---------------------------------------------------------------------------
__global__ __launch_bounds__(256) void attn_kernel(
    const _Float16* __restrict__ Qd, const _Float16* __restrict__ Kd,
    const _Float16* __restrict__ Vt, _Float16* __restrict__ Od) {
  __shared__ _Float16 Plds[8][16 * 32];   // 1KB / wave
  const int lane = threadIdx.x & 31;
  const int warp = threadIdx.x >> 5;
  const int wid  = blockIdx.x * 8 + warp;
  const int qt   = wid & 127;
  const int bh   = wid >> 7;              // b*12 + h
  const int n = lane & 15, hi = lane >> 4, m0 = hi * 8;

  const _Float16* Qbase = Qd + ((size_t)bh * NSEQ + qt * 16) * HD;
  const _Float16* Kbase = Kd + (size_t)bh * NSEQ * HD;
  const _Float16* Vbase = Vt + (size_t)bh * HD * NSEQ;

  const v16h qa0 = load_a_frag(Qbase, HD, 0, lane);
  const v16h qa1 = load_a_frag(Qbase, HD, 32, lane);

  float mrun[8], lrun[8];
  v8f o[4] = {};
#pragma unroll
  for (int r = 0; r < 8; ++r) { mrun[r] = -3.0e38f; lrun[r] = 0.f; }

  _Float16* P = &Plds[warp][0];

  for (int j = 0; j < NSEQ; j += 32) {
    v8f S0 = {}, S1 = {};
    {
      const _Float16* c0 = Kbase + (size_t)(j + n) * HD;
      S0 = wmma16(qa0, load_b_frag(c0, 0, hi), S0);
      S0 = wmma16(qa1, load_b_frag(c0, 32, hi), S0);
      const _Float16* c1 = Kbase + (size_t)(j + 16 + n) * HD;
      S1 = wmma16(qa0, load_b_frag(c1, 0, hi), S1);
      S1 = wmma16(qa1, load_b_frag(c1, 32, hi), S1);
    }
#pragma unroll
    for (int r = 0; r < 8; ++r) {
      float s0 = S0[r], s1 = S1[r];
      float mx = fmaxf(s0, s1);
      mx = fmaxf(mx, __shfl_xor(mx, 1, 16));
      mx = fmaxf(mx, __shfl_xor(mx, 2, 16));
      mx = fmaxf(mx, __shfl_xor(mx, 4, 16));
      mx = fmaxf(mx, __shfl_xor(mx, 8, 16));
      float mnew  = fmaxf(mrun[r], mx);
      float alpha = __expf(mrun[r] - mnew);
      mrun[r] = mnew;
      float p0 = __expf(s0 - mnew);
      float p1 = __expf(s1 - mnew);
      float rs = p0 + p1;
      rs += __shfl_xor(rs, 1, 16);
      rs += __shfl_xor(rs, 2, 16);
      rs += __shfl_xor(rs, 4, 16);
      rs += __shfl_xor(rs, 8, 16);
      lrun[r] = lrun[r] * alpha + rs;
#pragma unroll
      for (int t = 0; t < 4; ++t) o[t][r] *= alpha;
      int row = m0 + r;
      P[row * 32 + n]      = (_Float16)p0;
      P[row * 32 + 16 + n] = (_Float16)p1;
    }
    // In-wave LDS RAW (store above, ds_load below): DS ops are in-order per
    // wave, but wait DScnt explicitly before consuming.
    asm volatile("s_wait_dscnt 0" ::: "memory");
    v16h pf = load_a_frag(P, 32, 0, lane);   // P in A-layout (16x32)
#pragma unroll
    for (int t = 0; t < 4; ++t) {
      const _Float16* vcol = Vbase + (size_t)(t * 16 + n) * NSEQ + j;
      o[t] = wmma16(pf, load_b_frag(vcol, 0, hi), o[t]);
    }
  }

  const int b = bh / HEADS, h = bh % HEADS;
#pragma unroll
  for (int t = 0; t < 4; ++t) {
#pragma unroll
    for (int r = 0; r < 8; ++r) {
      int q = qt * 16 + m0 + r;
      float v = o[t][r] / lrun[r];
      Od[((size_t)(b * NSEQ + q)) * DIM + h * HD + t * 16 + n] = (_Float16)v;
    }
  }
}

// ---------------------------------------------------------------------------
// Stage 3: out = O @ W_proj + b_proj (f32 output). One wave -> 32x64 tile,
// fully-unrolled ping-pong K loop. 256 x 12 tiles = 3072 waves = 384 blocks.
// ---------------------------------------------------------------------------
__global__ __launch_bounds__(256) void proj_gemm_kernel(
    const _Float16* __restrict__ Oh, const _Float16* __restrict__ wt,
    const float* __restrict__ bias, float* __restrict__ out) {
  const int lane = threadIdx.x & 31;
  const int wid  = blockIdx.x * 8 + (threadIdx.x >> 5);
  const int rowTile = wid / 12;
  const int colT    = wid - rowTile * 12;
  const int n = lane & 15, hi = lane >> 4, m0 = hi * 8;

  const _Float16* arow0 = Oh + (size_t)rowTile * 32 * DIM;
  const _Float16* arow1 = arow0 + (size_t)16 * DIM;
  const _Float16* colbase[4];
#pragma unroll
  for (int t = 0; t < 4; ++t)
    colbase[t] = wt + (size_t)(colT * 64 + t * 16 + n) * DIM;

  v8f acc[2][4] = {};
  v16h a0[2], a1[2], b[2][4];
  a0[0] = load_a_frag(arow0, DIM, 0, lane);
  a1[0] = load_a_frag(arow1, DIM, 0, lane);
#pragma unroll
  for (int t = 0; t < 4; ++t) b[0][t] = load_b_frag(colbase[t], 0, hi);

#pragma unroll
  for (int kk = 0; kk < KSTEPS; ++kk) {
    const int cur = kk & 1, nx = cur ^ 1;
    if (kk + 1 < KSTEPS) {
      const int k0 = (kk + 1) * 32;
      a0[nx] = load_a_frag(arow0, DIM, k0, lane);
      a1[nx] = load_a_frag(arow1, DIM, k0, lane);
#pragma unroll
      for (int t = 0; t < 4; ++t) b[nx][t] = load_b_frag(colbase[t], k0, hi);
    }
#pragma unroll
    for (int t = 0; t < 4; ++t) {
      acc[0][t] = wmma16(a0[cur], b[cur][t], acc[0][t]);
      acc[1][t] = wmma16(a1[cur], b[cur][t], acc[1][t]);
    }
  }

#pragma unroll
  for (int g = 0; g < 2; ++g) {
#pragma unroll
    for (int t = 0; t < 4; ++t) {
#pragma unroll
      for (int r = 0; r < 8; ++r) {
        int c = colT * 64 + t * 16 + n;
        int R = rowTile * 32 + g * 16 + m0 + r;
        out[(size_t)R * DIM + c] = acc[g][t][r] + bias[c];
      }
    }
  }
}

// ---------------------------------------------------------------------------
extern "C" void kernel_launch(void* const* d_in, const int* in_sizes, int n_in,
                              void* d_out, int out_size, void* d_ws,
                              size_t ws_size, hipStream_t stream) {
  (void)in_sizes; (void)n_in; (void)out_size; (void)ws_size;
  const float* x     = (const float*)d_in[0];
  const float* Wqkv  = (const float*)d_in[1];
  const float* Wproj = (const float*)d_in[2];
  const float* bproj = (const float*)d_in[3];
  float* out = (float*)d_out;

  // Workspace carve-up (all sizes 256B-aligned). Total ~64.5 MB.
  char* ws = (char*)d_ws;
  _Float16* xh     = (_Float16*)ws; ws += (size_t)BN * DIM * 2;        // 12.6MB
  _Float16* wqkvT  = (_Float16*)ws; ws += (size_t)DIM * 3 * DIM * 2;   // 3.5MB
  _Float16* wprojT = (_Float16*)ws; ws += (size_t)DIM * DIM * 2;       // 1.2MB
  _Float16* Qd     = (_Float16*)ws; ws += (size_t)BN * DIM * 2;
  _Float16* Kd     = (_Float16*)ws; ws += (size_t)BN * DIM * 2;
  _Float16* Vt     = (_Float16*)ws; ws += (size_t)BN * DIM * 2;
  _Float16* Od     = (_Float16*)ws; ws += (size_t)BN * DIM * 2;

  const int nX = BN * DIM;                 // 6,291,456
  cvt_f32_to_f16<<<(nX + 255) / 256, 256, 0, stream>>>(x, xh, nX);
  transpose_to_f16<<<(DIM * 3 * DIM + 255) / 256, 256, 0, stream>>>(
      Wqkv, wqkvT, DIM, 3 * DIM);
  transpose_to_f16<<<(DIM * DIM + 255) / 256, 256, 0, stream>>>(
      Wproj, wprojT, DIM, DIM);

  qkv_gemm_kernel<<<1152, 256, 0, stream>>>(xh, wqkvT, Qd, Kd, Vt);
  attn_kernel<<<768, 256, 0, stream>>>(Qd, Kd, Vt, Od);
  proj_gemm_kernel<<<384, 256, 0, stream>>>(Od, wprojT, bproj, out);
}
